// DeepseekV3MoE_13778255085722
// MI455X (gfx1250) — matrix operand order repaired and checked by reading, not compiled, not run on hardware
//
#include <hip/hip_runtime.h>

#define TTOK  4096   // B*S tokens
#define HDIM  1024
#define NEXP  16
#define IDIM  512
#define ISDIM 1024

typedef __attribute__((ext_vector_type(16))) __bf16 v16bf;
typedef __attribute__((ext_vector_type(8)))  __bf16 v8bf;
typedef __attribute__((ext_vector_type(2)))  __bf16 v2bf;
typedef __attribute__((ext_vector_type(8)))  float  v8f;
typedef __attribute__((ext_vector_type(4)))  int    i4v;

#if defined(__gfx1250__) && __has_builtin(__builtin_amdgcn_global_load_async_to_lds_b128) && __has_builtin(__builtin_amdgcn_s_wait_asynccnt)
#define USE_ASYNC 1
#define GAS __attribute__((address_space(1)))
#define LAS __attribute__((address_space(3)))
// copy n16*16 bytes global(bf16 data) -> LDS via ASYNCcnt-tracked async DMA
__device__ __forceinline__ void async_cp16(const void* gsrc, void* ldst, int n16) {
    GAS i4v* g = (GAS i4v*)gsrc;
    LAS i4v* l = (LAS i4v*)ldst;
    for (int i = 0; i < n16; ++i)
        __builtin_amdgcn_global_load_async_to_lds_b128(g + i, l + i, 0, 0);
}
#else
#define USE_ASYNC 0
#endif

__device__ __forceinline__ unsigned short f2bf(float f) {
    unsigned int u = __float_as_uint(f);
    u += 0x7FFFu + ((u >> 16) & 1u);          // round-to-nearest-even
    return (unsigned short)(u >> 16);
}
// native packed bf16 convert: single v_cvt_pk_bf16_f32 when available
#if defined(__gfx1250__) && __has_builtin(__builtin_amdgcn_cvt_pk_bf16_f32)
__device__ __forceinline__ unsigned int pack2bf(float a, float b) {
    union { v2bf v; unsigned int u; } q;
    q.v = __builtin_amdgcn_cvt_pk_bf16_f32(a, b);
    return q.u;
}
#else
__device__ __forceinline__ unsigned int pack2bf(float a, float b) {
    union { __bf16 h[2]; unsigned int u; } q;
    q.h[0] = (__bf16)a; q.h[1] = (__bf16)b;
    return q.u;
}
#endif
__device__ __forceinline__ float sig_(float x) { return 1.0f / (1.0f + __expf(-x)); }

// ---------------------------------------------------------------- utilities
__global__ void k_zero_counts(int* counts) {
    if (threadIdx.x < NEXP) counts[threadIdx.x] = 0;
}

__global__ void k_offsets(const int* __restrict__ counts, int* __restrict__ offs) {
    if (threadIdx.x == 0) {
        int run = 0;
        for (int e = 0; e < NEXP; ++e) { offs[e] = run; run += counts[e]; }
    }
}

// fp32 -> bf16 bulk convert, 4 elements per thread
__global__ void k_cvt_bf16(const float* __restrict__ x, unsigned short* __restrict__ y) {
    int i = blockIdx.x * blockDim.x + threadIdx.x;          // over n/4
    float4 v = ((const float4*)x)[i];
    ((uint2*)y)[i] = make_uint2(pack2bf(v.x, v.y), pack2bf(v.z, v.w));
}

// ---------------------------------------------------------------- gate: sigmoid top-2 + dispatch lists
__global__ void k_gate(const float* __restrict__ x, const float* __restrict__ gw,
                       int* __restrict__ counts, int* __restrict__ tok_idx,
                       float* __restrict__ tok_w) {
    int t = blockIdx.x * blockDim.x + threadIdx.x;
    if (t >= TTOK) return;
    const float4* xr = (const float4*)(x + (size_t)t * HDIM);
    float acc[NEXP];
#pragma unroll
    for (int e = 0; e < NEXP; ++e) acc[e] = 0.0f;
    for (int h = 0; h < HDIM / 4; ++h) {
        float4 xv = xr[h];
#pragma unroll
        for (int e = 0; e < NEXP; ++e) {
            float4 g = ((const float4*)(gw + (size_t)e * HDIM))[h];
            acc[e] += xv.x * g.x + xv.y * g.y + xv.z * g.z + xv.w * g.w;
        }
    }
    float s[NEXP];
#pragma unroll
    for (int e = 0; e < NEXP; ++e) s[e] = sig_(acc[e]);
    int i0 = 0; float b0 = s[0];
#pragma unroll
    for (int e = 1; e < NEXP; ++e) if (s[e] > b0) { b0 = s[e]; i0 = e; }
    int i1 = (i0 == 0) ? 1 : 0; float b1 = s[i1];
#pragma unroll
    for (int e = 0; e < NEXP; ++e) if (e != i0 && s[e] > b1) { b1 = s[e]; i1 = e; }
    float inv = 1.0f / (b0 + b1 + 1e-20f);
    int s0 = atomicAdd(&counts[i0], 1);
    tok_idx[i0 * TTOK + s0] = t; tok_w[i0 * TTOK + s0] = b0 * inv;
    int s1 = atomicAdd(&counts[i1], 1);
    tok_idx[i1 * TTOK + s1] = t; tok_w[i1 * TTOK + s1] = b1 * inv;
}

// ---------------------------------------------------------------- stage 1: fused gate+up GEMM + SwiGLU
__global__ __launch_bounds__(256) void k_mlp_in(
    const unsigned short* __restrict__ xb,     // [TTOK, H] bf16
    const float* __restrict__ Wg_base,         // routed: [E,H,N]  shared: [H,N]
    const float* __restrict__ Wu_base,
    unsigned short* __restrict__ Hdst,         // [rows, N] bf16 (compact rows if routed)
    const int* __restrict__ counts, const int* __restrict__ offs,
    const int* __restrict__ tok_idx, int N, int routed)
{
    const int e  = routed ? blockIdx.z : 0;
    const int m0 = blockIdx.x * 128;
    const int n0 = blockIdx.y * 64;
    const int nrows = routed ? counts[e] : TTOK;
    if (m0 >= nrows) return;
    const int rowbase = routed ? offs[e] : 0;
    const float* Wg = Wg_base + (size_t)e * HDIM * N;
    const float* Wu = Wu_base + (size_t)e * HDIM * N;

    __shared__ int sTok[128];
    __shared__ __align__(16) unsigned short sA [128 * 32];   // [m][k]
    __shared__ __align__(16) unsigned short sBg[64 * 32];    // [n][k]
    __shared__ __align__(16) unsigned short sBu[64 * 32];

    const int tid  = threadIdx.x;
    const int lane = tid & 31;
    const int wv   = tid >> 5;

    if (tid < 128) {
        int r = m0 + tid;
        int tok = routed ? ((r < nrows) ? tok_idx[e * TTOK + r] : -1) : r;
        sTok[tid] = tok;
        if (tok < 0) {  // pre-zero invalid rows once (never rewritten)
            uint4 z = make_uint4(0,0,0,0);
            uint4* d = (uint4*)(sA + tid * 32);
            d[0] = z; d[1] = z; d[2] = z; d[3] = z;
        }
    }
    __syncthreads();

    v8f accG[4] = {}; v8f accU[4] = {};
    const int arow = tid >> 1;
    const int aseg = (tid & 1) * 16;
    const int half = lane >> 4;
    const int mrow = wv * 16 + (lane & 15);

    // hoist gathered A base address out of the K loop
    const unsigned short* asrc = nullptr;
    {
        int tok = sTok[arow];
        if (tok >= 0) asrc = xb + (size_t)tok * HDIM + aseg;
    }
    unsigned short* adst = sA + arow * 32 + aseg;

    for (int kk = 0; kk < HDIM; kk += 32) {
        // ---- A tile (gathered bf16 rows), 32B per thread
        if (asrc) {
#if USE_ASYNC
            async_cp16(asrc + kk, adst, 2);
#else
            uint4 p0 = ((const uint4*)(asrc + kk))[0];
            uint4 p1 = ((const uint4*)(asrc + kk))[1];
            ((uint4*)adst)[0] = p0; ((uint4*)adst)[1] = p1;
#endif
        }
        // ---- B tiles: fp32 -> packed bf16 pairs, transposed into [n][k]
#pragma unroll
        for (int j = 0; j < 4; ++j) {
            int lin = tid + j * 256;           // 0..1023
            int nc = lin & 63;                 // n within tile
            int kp = lin >> 6;                 // k-pair 0..15
            const float* pg = Wg + (size_t)(kk + 2 * kp) * N + n0 + nc;
            const float* pu = Wu + (size_t)(kk + 2 * kp) * N + n0 + nc;
            *(unsigned int*)(sBg + nc * 32 + kp * 2) = pack2bf(pg[0], pg[N]);
            *(unsigned int*)(sBu + nc * 32 + kp * 2) = pack2bf(pu[0], pu[N]);
        }
#if USE_ASYNC
        __builtin_amdgcn_s_wait_asynccnt(0);
#endif
        __syncthreads();

        // ---- fragments
        v8bf alo = *(const v8bf*)(sA + mrow * 32 + half * 16);
        v8bf ahi = *(const v8bf*)(sA + mrow * 32 + half * 16 + 8);
        v16bf a;
#pragma unroll
        for (int i = 0; i < 8; ++i) { a[i] = alo[i]; a[i + 8] = ahi[i]; }

#pragma unroll
        for (int ns = 0; ns < 4; ++ns) {
            int nr = ns * 16 + (lane & 15);
            v16bf bg = *(const v16bf*)(sBg + nr * 32 + half * 16);
            accG[ns] = __builtin_amdgcn_wmma_f32_16x16x32_bf16(false, a, false, bg, (short)0, accG[ns], false, false);
        }
#pragma unroll
        for (int ns = 0; ns < 4; ++ns) {
            int nr = ns * 16 + (lane & 15);
            v16bf bu = *(const v16bf*)(sBu + nr * 32 + half * 16);
            accU[ns] = __builtin_amdgcn_wmma_f32_16x16x32_bf16(false, a, false, bu, (short)0, accU[ns], false, false);
        }
        __syncthreads();
    }

    const int ncol = lane & 15;
#pragma unroll
    for (int v = 0; v < 8; ++v) {
        int ml = wv * 16 + half * 8 + v;
        if (m0 + ml < nrows) {
            size_t row = (size_t)(rowbase + m0 + ml);
#pragma unroll
            for (int ns = 0; ns < 4; ++ns) {
                float g = accG[ns][v], u = accU[ns][v];
                Hdst[row * N + n0 + ns * 16 + ncol] = f2bf(g * sig_(g) * u);
            }
        }
    }
}

// ---------------------------------------------------------------- stage 2: down-proj GEMM (+ scatter)
__global__ __launch_bounds__(256) void k_mlp_out(
    const unsigned short* __restrict__ Hsrc,   // [rows, K] bf16
    const float* __restrict__ Wd_base,         // routed: [E,K,H]  shared: [K,H]
    float* __restrict__ out,                   // [TTOK, H]
    const int* __restrict__ counts, const int* __restrict__ offs,
    const int* __restrict__ tok_idx, const float* __restrict__ tok_w,
    int K, int routed)
{
    const int e  = routed ? blockIdx.z : 0;
    const int m0 = blockIdx.x * 128;
    const int n0 = blockIdx.y * 64;
    const int nrows = routed ? counts[e] : TTOK;
    if (m0 >= nrows) return;
    const int rowbase = routed ? offs[e] : 0;
    const float* Wd = Wd_base + (size_t)e * K * HDIM;

    __shared__ __align__(16) unsigned short sA[128 * 32];
    __shared__ __align__(16) unsigned short sB[64 * 32];

    const int tid  = threadIdx.x;
    const int lane = tid & 31;
    const int wv   = tid >> 5;

    if (tid < 128 && m0 + tid >= nrows) {      // pre-zero tail rows once
        uint4 z = make_uint4(0,0,0,0);
        uint4* d = (uint4*)(sA + tid * 32);
        d[0] = z; d[1] = z; d[2] = z; d[3] = z;
    }
    __syncthreads();

    v8f acc[4] = {};
    const int arow = tid >> 1;
    const int aseg = (tid & 1) * 16;
    const int half = lane >> 4;
    const int mrow = wv * 16 + (lane & 15);

    const unsigned short* asrc =
        ((m0 + arow) < nrows) ? (Hsrc + (size_t)(rowbase + m0 + arow) * K + aseg) : nullptr;
    unsigned short* adst = sA + arow * 32 + aseg;

    for (int kk = 0; kk < K; kk += 32) {
        if (asrc) {
#if USE_ASYNC
            async_cp16(asrc + kk, adst, 2);
#else
            uint4 p0 = ((const uint4*)(asrc + kk))[0];
            uint4 p1 = ((const uint4*)(asrc + kk))[1];
            ((uint4*)adst)[0] = p0; ((uint4*)adst)[1] = p1;
#endif
        }
#pragma unroll
        for (int j = 0; j < 4; ++j) {
            int lin = tid + j * 256;
            int nc = lin & 63, kp = lin >> 6;
            const float* pd = Wd + (size_t)(kk + 2 * kp) * HDIM + n0 + nc;
            *(unsigned int*)(sB + nc * 32 + kp * 2) = pack2bf(pd[0], pd[HDIM]);
        }
#if USE_ASYNC
        __builtin_amdgcn_s_wait_asynccnt(0);
#endif
        __syncthreads();

        v8bf alo = *(const v8bf*)(sA + mrow * 32 + half * 16);
        v8bf ahi = *(const v8bf*)(sA + mrow * 32 + half * 16 + 8);
        v16bf a;
#pragma unroll
        for (int i = 0; i < 8; ++i) { a[i] = alo[i]; a[i + 8] = ahi[i]; }

#pragma unroll
        for (int ns = 0; ns < 4; ++ns) {
            int nr = ns * 16 + (lane & 15);
            v16bf b = *(const v16bf*)(sB + nr * 32 + half * 16);
            acc[ns] = __builtin_amdgcn_wmma_f32_16x16x32_bf16(false, a, false, b, (short)0, acc[ns], false, false);
        }
        __syncthreads();
    }

    const int ncol = lane & 15;
#pragma unroll
    for (int v = 0; v < 8; ++v) {
        int ml = wv * 16 + half * 8 + v;
        if (m0 + ml >= nrows) continue;
        if (routed) {
            int   tok = tok_idx[e * TTOK + m0 + ml];
            float wt  = tok_w [e * TTOK + m0 + ml];
#pragma unroll
            for (int ns = 0; ns < 4; ++ns)
                atomicAdd(&out[(size_t)tok * HDIM + n0 + ns * 16 + ncol], wt * acc[ns][v]);
        } else {
#pragma unroll
            for (int ns = 0; ns < 4; ++ns)
                out[(size_t)(m0 + ml) * HDIM + n0 + ns * 16 + ncol] = acc[ns][v];
        }
    }
}

// ---------------------------------------------------------------- host
extern "C" void kernel_launch(void* const* d_in, const int* in_sizes, int n_in,
                              void* d_out, int out_size, void* d_ws, size_t ws_size,
                              hipStream_t stream) {
    const float* x   = (const float*)d_in[0];  // [B,S,H]
    const float* gw  = (const float*)d_in[1];  // [E,H]
    const float* Weg = (const float*)d_in[2];  // [E,H,I]
    const float* Weu = (const float*)d_in[3];  // [E,H,I]
    const float* Wed = (const float*)d_in[4];  // [E,I,H]
    const float* Wsg = (const float*)d_in[5];  // [H,IS]
    const float* Wsu = (const float*)d_in[6];  // [H,IS]
    const float* Wsd = (const float*)d_in[7];  // [IS,H]
    float* out = (float*)d_out;

    // workspace carve (16B-aligned): ~24.6 MB total
    unsigned char* w = (unsigned char*)d_ws;
    int*   counts  = (int*)w;            w += 256;
    int*   offs    = (int*)w;            w += 256;
    int*   tok_idx = (int*)w;            w += (size_t)NEXP * TTOK * 4;   // 256 KB
    float* tok_w   = (float*)w;          w += (size_t)NEXP * TTOK * 4;   // 256 KB
    unsigned short* xb      = (unsigned short*)w; w += (size_t)TTOK * HDIM * 2;       // 8 MB
    unsigned short* Hroute  = (unsigned short*)w; w += (size_t)TTOK * 2 * IDIM * 2;   // 8 MB
    unsigned short* Hshared = (unsigned short*)w; w += (size_t)TTOK * ISDIM * 2;      // 8 MB

    k_zero_counts<<<1, 32, 0, stream>>>(counts);
    k_cvt_bf16<<<(TTOK * HDIM / 4) / 256, 256, 0, stream>>>(x, xb);
    k_gate<<<TTOK / 256, 256, 0, stream>>>(x, gw, counts, tok_idx, tok_w);
    k_offsets<<<1, 32, 0, stream>>>(counts, offs);

    // shared expert, stage 1
    k_mlp_in<<<dim3(32, ISDIM / 64, 1), 256, 0, stream>>>(
        xb, Wsg, Wsu, Hshared, counts, offs, tok_idx, ISDIM, 0);
    // routed experts, stage 1
    k_mlp_in<<<dim3(32, IDIM / 64, NEXP), 256, 0, stream>>>(
        xb, Weg, Weu, Hroute, counts, offs, tok_idx, IDIM, 1);

    // shared expert, stage 2: full overwrite of out (clean across graph replays)
    k_mlp_out<<<dim3(32, HDIM / 64, 1), 256, 0, stream>>>(
        Hshared, Wsd, out, counts, offs, tok_idx, tok_w, ISDIM, 0);
    // routed experts, stage 2: weighted atomic scatter-add
    k_mlp_out<<<dim3(32, HDIM / 64, NEXP), 256, 0, stream>>>(
        Hroute, Wed, out, counts, offs, tok_idx, tok_w, IDIM, 1);
}
